// NonLocalBlockND_9242769621741
// MI455X (gfx1250) — compile-verified, matrix operand order checked by reading, not run on hardware
//
#include <hip/hip_runtime.h>
#include <hip/hip_bf16.h>

// ---------------------------------------------------------------------------
// NonLocalBlockND on MI455X (gfx1250, wave32, WMMA f32_16x16x32_f16)
// B=16 C=256 CI=128 H=W=64  N=4096  M=1024
// ---------------------------------------------------------------------------

#define B_   16
#define C_   256
#define CI_  128
#define N_   4096   // H*W
#define M_   1024   // (H/2)*(W/2)
#define EPSb 1e-5f

typedef _Float16 h2_t   __attribute__((ext_vector_type(2)));
typedef _Float16 v8h_t  __attribute__((ext_vector_type(8)));
typedef _Float16 v16h_t __attribute__((ext_vector_type(16)));
typedef float    v8f_t  __attribute__((ext_vector_type(8)));
typedef __fp16   fp16x2 __attribute__((ext_vector_type(2)));

union U16h { v16h_t v; v8h_t h[2]; h2_t p[8]; };
union U8h  { v8h_t  v; h2_t p[4]; };

__device__ __forceinline__ h2_t pkrtz(float a, float b) {
  fp16x2 r = __builtin_amdgcn_cvt_pkrtz(a, b);
  return __builtin_bit_cast(h2_t, r);
}

__device__ __forceinline__ v8f_t vzero8() {
  v8f_t z = {0.f,0.f,0.f,0.f,0.f,0.f,0.f,0.f};
  return z;
}

#define WMMA_F16(a,b,c) \
  __builtin_amdgcn_wmma_f32_16x16x32_f16(false,(a),false,(b),(short)0,(c),false,false)

// A/B operand (16x32 f16, K-major rows) from f16 memory.
// lane 0-15 : K = 0..7 , 16..23   (VGPR0-3, VGPR4-7)
// lane16-31 : K = 8..15, 24..31
__device__ __forceinline__ v16h_t load_ab_f16(const _Float16* rowK) {
  const int hi = (threadIdx.x & 16) ? 8 : 0;
  U16h u;
  u.h[0] = *(const v8h_t*)(rowK + hi);
  u.h[1] = *(const v8h_t*)(rowK + hi + 16);
  return u.v;
}

// Same operand but sourced from f32 memory (convert+pack rtz).
__device__ __forceinline__ v16h_t load_ab_f32(const float* rowK) {
  const int hi = (threadIdx.x & 16) ? 8 : 0;
  U16h u;
#pragma unroll
  for (int cc = 0; cc < 2; ++cc) {
    const float* p = rowK + hi + cc * 16;
    float4 f0 = *(const float4*)(p);
    float4 f1 = *(const float4*)(p + 4);
    u.p[cc*4+0] = pkrtz(f0.x, f0.y);
    u.p[cc*4+1] = pkrtz(f0.z, f0.w);
    u.p[cc*4+2] = pkrtz(f1.x, f1.y);
    u.p[cc*4+3] = pkrtz(f1.z, f1.w);
  }
  return u.v;
}

__device__ __forceinline__ float xor16_max(float v) {
  float o = __int_as_float(__builtin_amdgcn_ds_swizzle(__float_as_int(v), 0x401F));
  return fmaxf(v, o);
}
__device__ __forceinline__ float xor16_add(float v) {
  float o = __int_as_float(__builtin_amdgcn_ds_swizzle(__float_as_int(v), 0x401F));
  return v + o;
}

// ===========================================================================
// Kernel 1: fused theta/phi/g 1x1-conv projections (+2x2 maxpool for phi/g).
//   out^T GEMM:  D[ci, n] = W[ci, c] * x[c, n]   (K = C = 256)
// Grid: 16 batches * 32 row-pairs. Block: 384 threads = 12 waves
// (waves 0-3: theta, 4-7: phi, 8-11: g; each sub-wave owns 32 ci rows).
// Outputs (f16):  th[b*N+n][CI]   ph[b*M+m][CI]   gx[b*CI+ci][M]
// ===========================================================================
#define K1_THREADS 384
#define K1_XP 40      // x-tile LDS pitch (halves): 80B, 16B-aligned rows
#define K1_SP 132     // staging pitch (floats)
#define K1_SHMEM (128*K1_XP*2 + 128*K1_SP*4)   // 10240 + 67584 = 77824 B

__global__ void __launch_bounds__(K1_THREADS)
proj_kernel(const float* __restrict__ x,
            const float* __restrict__ gw, const float* __restrict__ gb,
            const float* __restrict__ tw, const float* __restrict__ tb,
            const float* __restrict__ pw, const float* __restrict__ pb,
            _Float16* __restrict__ th, _Float16* __restrict__ ph,
            _Float16* __restrict__ gx) {
  extern __shared__ unsigned char smem[];
  _Float16* xt  = (_Float16*)smem;                       // [128 n][K1_XP c]
  float*    stg = (float*)(smem + 128 * K1_XP * 2);      // [128 ci][K1_SP n]

  const int tid   = threadIdx.x;
  const int wave  = tid >> 5;
  const int lane  = tid & 31;
  const int l15   = lane & 15;
  const int lhalf = lane >> 4;
  const int proj  = wave >> 2;    // 0 theta, 1 phi, 2 g
  const int q     = wave & 3;     // 32-ci-row slice

  const int b  = blockIdx.x >> 5;
  const int rp = blockIdx.x & 31;   // row-pair: image rows 2rp, 2rp+1
  const int p0 = rp * 128;          // pixel base within image

  const float* wsel = (proj == 0) ? tw : (proj == 1) ? pw : gw;

  v8f_t acc[2][8];
#pragma unroll
  for (int rt = 0; rt < 2; ++rt)
#pragma unroll
    for (int ct = 0; ct < 8; ++ct) acc[rt][ct] = vzero8();

  for (int kc = 0; kc < 8; ++kc) {          // K slabs of 32 channels
    __syncthreads();
    // cooperatively stage x slab [32 c][128 n] -> LDS transposed [n][c] f16
    for (int idx = tid; idx < 1024; idx += K1_THREADS) {
      int cl = idx >> 5;
      int n4 = (idx & 31) * 4;
      const float* px = x + (((size_t)b * C_ + kc * 32 + cl) * N_) + p0 + n4;
      float4 f = *(const float4*)px;
      h2_t a0 = pkrtz(f.x, f.y);
      h2_t a1 = pkrtz(f.z, f.w);
      xt[(n4+0)*K1_XP + cl] = a0[0];
      xt[(n4+1)*K1_XP + cl] = a0[1];
      xt[(n4+2)*K1_XP + cl] = a1[0];
      xt[(n4+3)*K1_XP + cl] = a1[1];
    }
    __syncthreads();

    v16h_t Aop[2];
#pragma unroll
    for (int rt = 0; rt < 2; ++rt) {
      int ci = q * 32 + rt * 16 + l15;
      Aop[rt] = load_ab_f32(wsel + (size_t)ci * C_ + kc * 32);
    }
#pragma unroll
    for (int ct = 0; ct < 8; ++ct) {
      int n = ct * 16 + l15;
      v16h_t Bop = load_ab_f16(xt + n * K1_XP);
      acc[0][ct] = WMMA_F16(Aop[0], Bop, acc[0][ct]);
      acc[1][ct] = WMMA_F16(Aop[1], Bop, acc[1][ct]);
    }
  }

  // --- epilogues: dump D tiles to staging, then layout-specific stores ---
  for (int pe = 0; pe < 3; ++pe) {
    __syncthreads();
    if (proj == pe) {
#pragma unroll
      for (int rt = 0; rt < 2; ++rt)
#pragma unroll
        for (int ct = 0; ct < 8; ++ct)
#pragma unroll
          for (int r = 0; r < 8; ++r) {
            int ci = q * 32 + rt * 16 + r + lhalf * 8;   // D row
            int n  = ct * 16 + l15;                      // D col
            stg[ci * K1_SP + n] = acc[rt][ct][r];
          }
    }
    __syncthreads();
    const float* bias = (pe == 0) ? tb : (pe == 1) ? pb : gb;
    if (pe == 0) {
      // theta: [b*N + p0+n][CI] f16 rows
      for (int idx = tid; idx < 2048; idx += K1_THREADS) {
        int n  = idx >> 4;
        int c8 = (idx & 15) * 8;
        U8h u;
#pragma unroll
        for (int j = 0; j < 4; ++j) {
          float v0 = stg[(c8 + 2*j + 0) * K1_SP + n] + bias[c8 + 2*j + 0];
          float v1 = stg[(c8 + 2*j + 1) * K1_SP + n] + bias[c8 + 2*j + 1];
          u.p[j] = pkrtz(v0, v1);
        }
        *(v8h_t*)(th + ((size_t)b * N_ + p0 + n) * CI_ + c8) = u.v;
      }
    } else if (pe == 1) {
      // phi: pooled, [b*M + m][CI]
      for (int idx = tid; idx < 4096; idx += K1_THREADS) {
        int m  = idx >> 7;
        int ci = idx & 127;
        int n0 = 2 * m;
        float v = fmaxf(fmaxf(stg[ci*K1_SP + n0],      stg[ci*K1_SP + n0 + 1]),
                        fmaxf(stg[ci*K1_SP + n0 + 64], stg[ci*K1_SP + n0 + 65]));
        v += bias[ci];
        ph[((size_t)b * M_ + rp * 32 + m) * CI_ + ci] = (_Float16)v;
      }
    } else {
      // g: pooled, transposed [b*CI + ci][M]
      for (int idx = tid; idx < 4096; idx += K1_THREADS) {
        int ci = idx >> 5;
        int m  = idx & 31;
        int n0 = 2 * m;
        float v = fmaxf(fmaxf(stg[ci*K1_SP + n0],      stg[ci*K1_SP + n0 + 1]),
                        fmaxf(stg[ci*K1_SP + n0 + 64], stg[ci*K1_SP + n0 + 65]));
        v += bias[ci];
        gx[((size_t)b * CI_ + ci) * M_ + rp * 32 + m] = (_Float16)v;
      }
    }
  }
}

// ===========================================================================
// Kernel 2: fused attention.  Per WG: one batch, 64 query rows n.
//  Phase 1: S^T[m, n] WMMA over all M=1024, f16 stash in LDS, row max.
//  Phase 2: in-LDS softmax (exp + sum), keep P f16.
//  Phase 3: y[n, ci] = P @ g  WMMA (A read straight from stash rows).
// Block: 128 threads = 4 waves, wave w owns rows n_loc = 16w..16w+15.
// ===========================================================================
#define K2_THREADS 128
#define K2_PITCH   1032    // stash pitch in halves (2064B rows, 16B aligned)
#define K2_SHMEM   (64*K2_PITCH*2 + 64*4*2)   // 132096 + 512 = 132608 B

__global__ void __launch_bounds__(K2_THREADS)
attn_kernel(const _Float16* __restrict__ th, const _Float16* __restrict__ ph,
            const _Float16* __restrict__ gx, _Float16* __restrict__ y) {
  extern __shared__ unsigned char smem[];
  _Float16* stash = (_Float16*)smem;                    // [64 n][K2_PITCH m]
  float*    maxs  = (float*)(smem + 64 * K2_PITCH * 2); // [64]
  float*    rsum  = maxs + 64;                          // [64]

  const int tid   = threadIdx.x;
  const int wave  = tid >> 5;
  const int lane  = tid & 31;
  const int l15   = lane & 15;
  const int lhalf = lane >> 4;

  const int b     = blockIdx.x >> 6;
  const int nblk  = blockIdx.x & 63;
  const int nloc0 = wave * 16;
  const size_t thBase = ((size_t)b * N_ + nblk * 64) * CI_;
  const size_t phBase = (size_t)b * M_ * CI_;

  // ---- Phase 1: S^T tiles; B (theta) operands are loop-invariant ----
  v16h_t Bop[4];
  {
    int n = nloc0 + l15;
    for (int kt = 0; kt < 4; ++kt)
      Bop[kt] = load_ab_f16(th + thBase + (size_t)n * CI_ + kt * 32);
  }
  float maxv = -3.0e38f;
  for (int mt = 0; mt < 64; ++mt) {
    v8f_t acc = vzero8();
    int m = mt * 16 + l15;
#pragma unroll
    for (int kt = 0; kt < 4; ++kt) {
      v16h_t Aop = load_ab_f16(ph + phBase + (size_t)m * CI_ + kt * 32);
      acc = WMMA_F16(Aop, Bop[kt], acc);
    }
    int nl    = nloc0 + l15;           // D col -> query row
    int mbase = mt * 16 + lhalf * 8;   // D rows r(+8*half) -> key index m
#pragma unroll
    for (int r = 0; r < 8; r += 2) {
      maxv = fmaxf(maxv, fmaxf(acc[r], acc[r+1]));
      *(h2_t*)(stash + nl * K2_PITCH + mbase + r) = pkrtz(acc[r], acc[r+1]);
    }
  }
  maxv = xor16_max(maxv);
  if (lane < 16) maxs[nloc0 + lane] = maxv;
  __syncthreads();

  // ---- Phase 2: softmax over the stash (each lane: 512 keys of one row) --
  {
    int row = nloc0 + l15;
    float rmax = maxs[row];
    float s = 0.f;
    _Float16* prow = stash + row * K2_PITCH + lhalf * 512;
    for (int i = 0; i < 64; ++i) {
      v8h_t v = *(v8h_t*)(prow + i * 8);
#pragma unroll
      for (int j = 0; j < 8; ++j) {
        float ev = __expf((float)v[j] - rmax);
        s += ev;
        v[j] = (_Float16)ev;
      }
      *(v8h_t*)(prow + i * 8) = v;
    }
    s = xor16_add(s);
    if (lane < 16) rsum[row] = 1.0f / s;
  }
  __syncthreads();

  // ---- Phase 3: y = P @ g ----
  v8f_t acc[8];
#pragma unroll
  for (int ct = 0; ct < 8; ++ct) acc[ct] = vzero8();
  const int nrow = nloc0 + l15;
  for (int ks = 0; ks < 32; ++ks) {
    v16h_t Aop = load_ab_f16(stash + nrow * K2_PITCH + ks * 32);
#pragma unroll
    for (int ct = 0; ct < 8; ++ct) {
      int ci = ct * 16 + l15;
      v16h_t Bg = load_ab_f16(gx + ((size_t)b * CI_ + ci) * M_ + ks * 32);
      acc[ct] = WMMA_F16(Aop, Bg, acc[ct]);
    }
  }
  // normalize rows and store y f16 [b*N + n][CI]
#pragma unroll
  for (int r = 0; r < 8; ++r) {
    float rs = rsum[nloc0 + r + lhalf * 8];
    size_t base = ((size_t)b * N_ + nblk * 64 + nloc0 + r + lhalf * 8) * CI_;
#pragma unroll
    for (int ct = 0; ct < 8; ++ct)
      y[base + ct * 16 + l15] = (_Float16)(acc[ct][r] * rs);
  }
}

// ===========================================================================
// Kernel 3: z = y @ W^T + b  ([65536 x 128] x [128 x 256]) + BN partials.
// Block: 256 threads = 8 waves, 128 rows per WG (512 WGs).
// Writes z transposed: zt[c][b*N + n]  (coalesced for the apply kernel),
// plus per-WG per-channel sum / sumsq partials (no atomics -> deterministic).
// ===========================================================================
#define K3_THREADS 256
#define K3_WP 136     // staged-weight pitch (halves): 272B
#define K3_SP 260     // z staging pitch (floats)
#define K3_SHMEM (256*K3_WP*2 + 128*K3_SP*4)  // 69632 + 133120 = 202752 B

__global__ void __launch_bounds__(K3_THREADS)
outconv_kernel(const _Float16* __restrict__ y, const float* __restrict__ ww,
               const float* __restrict__ wb, float* __restrict__ zt,
               float* __restrict__ psum, float* __restrict__ psq) {
  extern __shared__ unsigned char smem[];
  _Float16* wl  = (_Float16*)smem;                    // [256 c][K3_WP ci]
  float*    stg = (float*)(smem + 256 * K3_WP * 2);   // [128 n][K3_SP c]

  const int tid   = threadIdx.x;
  const int wave  = tid >> 5;
  const int lane  = tid & 31;
  const int l15   = lane & 15;
  const int lhalf = lane >> 4;
  const int blk   = blockIdx.x;

  // stage W (f32 -> f16) into LDS
  for (int idx = tid; idx < 4096; idx += K3_THREADS) {
    int c  = idx >> 4;
    int g8 = (idx & 15) * 8;
    const float* p = ww + (size_t)c * CI_ + g8;
    float4 f0 = *(const float4*)p;
    float4 f1 = *(const float4*)(p + 4);
    U8h u;
    u.p[0] = pkrtz(f0.x, f0.y);
    u.p[1] = pkrtz(f0.z, f0.w);
    u.p[2] = pkrtz(f1.x, f1.y);
    u.p[3] = pkrtz(f1.z, f1.w);
    *(v8h_t*)(wl + c * K3_WP + g8) = u.v;
  }
  __syncthreads();

  v8f_t acc[16];
#pragma unroll
  for (int ct = 0; ct < 16; ++ct) acc[ct] = vzero8();

  const size_t yrow = ((size_t)blk * 128 + wave * 16 + l15) * CI_;
#pragma unroll
  for (int kt = 0; kt < 4; ++kt) {
    v16h_t Aop = load_ab_f16(y + yrow + kt * 32);
#pragma unroll
    for (int ct = 0; ct < 16; ++ct) {
      v16h_t Bop = load_ab_f16(wl + (ct * 16 + l15) * K3_WP + kt * 32);
      acc[ct] = WMMA_F16(Aop, Bop, acc[ct]);
    }
  }

  // bias + dump to LDS staging [n][c]
#pragma unroll
  for (int ct = 0; ct < 16; ++ct) {
    int c = ct * 16 + l15;
    float bias = wb[c];
#pragma unroll
    for (int r = 0; r < 8; ++r) {
      int nl = wave * 16 + r + lhalf * 8;
      stg[nl * K3_SP + c] = acc[ct][r] + bias;
    }
  }
  __syncthreads();

  // per-channel partial sums + transposed global store (thread == channel)
  {
    int c = tid;
    float s = 0.f, q = 0.f;
    for (int nl = 0; nl < 128; nl += 4) {
      float4 v;
      v.x = stg[(nl + 0) * K3_SP + c];
      v.y = stg[(nl + 1) * K3_SP + c];
      v.z = stg[(nl + 2) * K3_SP + c];
      v.w = stg[(nl + 3) * K3_SP + c];
      s += v.x + v.y + v.z + v.w;
      q += v.x*v.x + v.y*v.y + v.z*v.z + v.w*v.w;
      *(float4*)(zt + (size_t)c * (B_ * N_) + blk * 128 + nl) = v;
    }
    psum[blk * C_ + c] = s;
    psq [blk * C_ + c] = q;
  }
}

// ===========================================================================
// Kernel 4: finalize BN statistics -> per-channel scale/shift.
// ===========================================================================
__global__ void __launch_bounds__(256)
bnstats_kernel(const float* __restrict__ psum, const float* __restrict__ psq,
               const float* __restrict__ gamma, const float* __restrict__ beta,
               float* __restrict__ scale, float* __restrict__ shift) {
  int c = threadIdx.x;
  float s = 0.f, q = 0.f;
  for (int blkI = 0; blkI < 512; ++blkI) {
    s += psum[blkI * C_ + c];
    q += psq [blkI * C_ + c];
  }
  const float invN = 1.0f / (float)(B_ * N_);
  float mean = s * invN;
  float var  = q * invN - mean * mean;      // biased variance
  float rstd = rsqrtf(var + EPSb);
  float sc   = gamma[c] * rstd;
  scale[c] = sc;
  shift[c] = beta[c] - mean * sc;
}

// ===========================================================================
// Kernel 5: out[b,c,:] = zt[c, b*N + :]*scale[c] + shift[c] + x[b,c,:]
// ===========================================================================
__global__ void __launch_bounds__(256)
apply_kernel(const float* __restrict__ zt, const float* __restrict__ x,
             const float* __restrict__ scale, const float* __restrict__ shift,
             float* __restrict__ out) {
  int b = blockIdx.x >> 8;
  int c = blockIdx.x & 255;
  float sc = scale[c], sh = shift[c];
  const float* zr = zt + (size_t)c * (B_ * N_) + (size_t)b * N_;
  const float* xr = x  + ((size_t)b * C_ + c) * N_;
  float*       op = out + ((size_t)b * C_ + c) * N_;
  for (int i = threadIdx.x; i < N_ / 4; i += 256) {
    float4 z4 = *(const float4*)(zr + i * 4);
    float4 x4 = *(const float4*)(xr + i * 4);
    float4 o;
    o.x = z4.x * sc + sh + x4.x;
    o.y = z4.y * sc + sh + x4.y;
    o.z = z4.z * sc + sh + x4.z;
    o.w = z4.w * sc + sh + x4.w;
    *(float4*)(op + i * 4) = o;
  }
}

// ===========================================================================
// Launch
// ===========================================================================
extern "C" void kernel_launch(void* const* d_in, const int* in_sizes, int n_in,
                              void* d_out, int out_size, void* d_ws, size_t ws_size,
                              hipStream_t stream) {
  const float* x   = (const float*)d_in[0];
  const float* g_w = (const float*)d_in[1];
  const float* g_b = (const float*)d_in[2];
  const float* t_w = (const float*)d_in[3];
  const float* t_b = (const float*)d_in[4];
  const float* p_w = (const float*)d_in[5];
  const float* p_b = (const float*)d_in[6];
  const float* w_w = (const float*)d_in[7];
  const float* w_b = (const float*)d_in[8];
  const float* bng = (const float*)d_in[9];
  const float* bnb = (const float*)d_in[10];
  float* out = (float*)d_out;
  (void)in_sizes; (void)n_in; (void)out_size; (void)ws_size;

  unsigned char* ws = (unsigned char*)d_ws;
  // workspace layout (bytes), total ~110.2 MB
  const size_t OFF_TH  = 0;                                   // f16 [B*N][CI]   16 MB
  const size_t OFF_PH  = OFF_TH + (size_t)B_*N_*CI_*2;        // f16 [B*M][CI]    4 MB
  const size_t OFF_G   = OFF_PH + (size_t)B_*M_*CI_*2;        // f16 [B*CI][M]    4 MB
  const size_t OFF_Y   = OFF_G  + (size_t)B_*CI_*M_*2;        // f16 [B*N][CI]   16 MB
  const size_t OFF_ZT  = OFF_Y  + (size_t)B_*N_*CI_*2;        // f32 [C][B*N]    64 MB
  const size_t OFF_PS  = OFF_ZT + (size_t)C_*B_*N_*4;         // f32 [512][C]   0.5 MB
  const size_t OFF_PQ  = OFF_PS + (size_t)512*C_*4;           // f32 [512][C]   0.5 MB
  const size_t OFF_SC  = OFF_PQ + (size_t)512*C_*4;           // f32 [C]
  const size_t OFF_SH  = OFF_SC + (size_t)C_*4;               // f32 [C]

  _Float16* th = (_Float16*)(ws + OFF_TH);
  _Float16* ph = (_Float16*)(ws + OFF_PH);
  _Float16* gx = (_Float16*)(ws + OFF_G);
  _Float16* yb = (_Float16*)(ws + OFF_Y);
  float* zt    = (float*)(ws + OFF_ZT);
  float* psum  = (float*)(ws + OFF_PS);
  float* psq   = (float*)(ws + OFF_PQ);
  float* scale = (float*)(ws + OFF_SC);
  float* shift = (float*)(ws + OFF_SH);

  proj_kernel<<<B_ * 32, K1_THREADS, K1_SHMEM, stream>>>(
      x, g_w, g_b, t_w, t_b, p_w, p_b, th, ph, gx);

  attn_kernel<<<B_ * 64, K2_THREADS, K2_SHMEM, stream>>>(th, ph, gx, yb);

  outconv_kernel<<<512, K3_THREADS, K3_SHMEM, stream>>>(yb, w_w, w_b, zt, psum, psq);

  bnstats_kernel<<<1, 256, 0, stream>>>(psum, psq, bng, bnb, scale, shift);

  apply_kernel<<<B_ * C_, 256, 0, stream>>>(zt, x, scale, shift, out);
}
